// ComplexEagerAttentionModel_44693429682908
// MI455X (gfx1250) — matrix operand order repaired and checked by reading, not compiled, not run on hardware
//
#include <hip/hip_runtime.h>

// ---------------------------------------------------------------------------
// MHA block on gfx1250 (MI455X): bf16 WMMA everywhere, f32 accumulation.
// B=2, S=2048, D=2048, H=16, HD=128, M = B*S = 4096.
// Pass 0: convert x + all weights to bf16 (once).
// Pass 1-3: QKV GEMMs (bf16 in, bf16 out; V stored transposed).
// Pass 4: flash attention (1 wave / 16-row q tile).
// Pass 5: output projection GEMM (bf16 in, f32 out).
// ---------------------------------------------------------------------------

typedef __bf16 bf16_t;
typedef __attribute__((ext_vector_type(16))) __bf16 v16bf;
typedef __attribute__((ext_vector_type(8)))  float  v8f;
typedef __attribute__((ext_vector_type(4)))  int    v4i;

#define DIM_S   2048
#define DIM_D   2048
#define DIM_H   16
#define DIM_HD  128
#define DIM_M   4096          // B*S
#define ATT_SCALE 0.08838834764831845f   // 1/sqrt(128)
#define NEG_BIG  (-3.0e38f)

// ---- CDNA5 async global->LDS path (guarded; sync fallback) ----------------
#if defined(__has_builtin)
#  if __has_builtin(__builtin_amdgcn_global_load_async_to_lds_b128)
#    define HAVE_ASYNC_LDS 1
#  endif
#endif
#ifndef HAVE_ASYNC_LDS
#  define HAVE_ASYNC_LDS 0
#endif

typedef __attribute__((address_space(1))) v4i* g_v4i_p;   // global int4*
typedef __attribute__((address_space(3))) v4i* l_v4i_p;   // LDS int4*

static __device__ __forceinline__ void copy16_to_lds(const void* g, void* l) {
#if HAVE_ASYNC_LDS
    __builtin_amdgcn_global_load_async_to_lds_b128((g_v4i_p)g, (l_v4i_p)l, 0, 0);
#else
    *(uint4*)l = *(const uint4*)g;
#endif
}

static __device__ __forceinline__ void wait_lds_copies() {
#if HAVE_ASYNC_LDS
#  if defined(__has_builtin) && __has_builtin(__builtin_amdgcn_s_wait_asynccnt)
    __builtin_amdgcn_s_wait_asynccnt(0);
#  else
    asm volatile("s_wait_asynccnt 0x0" ::: "memory");
#  endif
#endif
}

// ---------------------------------------------------------------------------
struct BF16Frag { uint4 lo, hi; };

static __device__ __forceinline__ v16bf make_frag(uint4 lo, uint4 hi) {
    BF16Frag t{lo, hi};
    return __builtin_bit_cast(v16bf, t);
}

// round-to-nearest-even f32 -> bf16 (as raw u16)
static __device__ __forceinline__ unsigned short f32_bf16(float f) {
    unsigned int u = __float_as_uint(f);
    unsigned int r = u + 0x7FFFu + ((u >> 16) & 1u);
    return (unsigned short)(r >> 16);
}

// ---------------------------------------------------------------------------
// Elementwise f32 -> packed bf16 (8 elements / thread).
// ---------------------------------------------------------------------------
__global__ __launch_bounds__(256)
void cvt_bf16_kernel(const float* __restrict__ in, unsigned short* __restrict__ out, int n8)
{
    const int i = blockIdx.x * 256 + threadIdx.x;
    if (i >= n8) return;
    const float4* g = (const float4*)(in + (size_t)i * 8);
    float4 a = g[0], b = g[1];
    uint4 p;
    p.x = f32_bf16(a.x) | ((unsigned)f32_bf16(a.y) << 16);
    p.y = f32_bf16(a.z) | ((unsigned)f32_bf16(a.w) << 16);
    p.z = f32_bf16(b.x) | ((unsigned)f32_bf16(b.y) << 16);
    p.w = f32_bf16(b.z) | ((unsigned)f32_bf16(b.w) << 16);
    ((uint4*)out)[i] = p;
}

// ---------------------------------------------------------------------------
// GEMM: C[m,n] = A[m,:] . W[n,:] + bias[n], A/W bf16 row-major [*,2048].
// MODE 0: store f32 row-major [M, D]           (output projection)
// MODE 1: store bf16 at [b, h, s, hd]          (Q, K)
// MODE 2: store bf16 at [b, h, hd, s]          (V transposed)
// Block: 256 threads = 8 waves, 128x64 output tile, K-step 32.
// Each wave: 32x32 sub-block = 4 WMMAs per K-step.
// ---------------------------------------------------------------------------
template <int MODE>
__global__ __launch_bounds__(256)
void gemm_bias_kernel(const unsigned short* __restrict__ A,
                      const unsigned short* __restrict__ W,
                      const float* __restrict__ bias, void* __restrict__ out)
{
    __shared__ unsigned short lx[128 * 32];   // A tile  (8 KB)
    __shared__ unsigned short lw[64 * 32];    // W tile  (4 KB)

    const int tid   = threadIdx.x;
    const int lane  = tid & 31;
    const int wave  = tid >> 5;
    const int hl    = lane >> 4;       // half-wave select
    const int l16   = lane & 15;
    const int tileN = blockIdx.x * 64;
    const int tileM = blockIdx.y * 128;

    const int mi0 = (wave & 3) * 2;    // two 16-row subtiles
    const int nj0 = (wave >> 2) * 2;   // two 16-col subtiles

    v8f acc[2][2];
    acc[0][0] = v8f{}; acc[0][1] = v8f{};
    acc[1][0] = v8f{}; acc[1][1] = v8f{};

    // staging map: 16B chunks; X: 128 rows x 4 chunks = 512 (2/thread),
    // W: 64 rows x 4 chunks = 256 (1/thread)
    const int xr0 = tid >> 2,          xs0 = (tid & 3) * 8;
    const int xr1 = (tid + 256) >> 2,  xs1 = xs0;            // same seg pattern
    const int wr  = tid >> 2,          wsg = (tid & 3) * 8;

    for (int k0 = 0; k0 < DIM_D; k0 += 32) {
        copy16_to_lds(A + (size_t)(tileM + xr0) * DIM_D + k0 + xs0, &lx[xr0 * 32 + xs0]);
        copy16_to_lds(A + (size_t)(tileM + xr1) * DIM_D + k0 + xs1, &lx[xr1 * 32 + xs1]);
        copy16_to_lds(W + (size_t)(tileN + wr)  * DIM_D + k0 + wsg, &lw[wr  * 32 + wsg]);
        wait_lds_copies();
        __syncthreads();

        v16bf af[2], bfr[2];
        #pragma unroll
        for (int i = 0; i < 2; ++i) {
            // A fragment: lane<16 holds K{0..7,16..23}; lane>=16 K{8..15,24..31}
            const uint4* ap = (const uint4*)&lx[((mi0 + i) * 16 + l16) * 32];
            af[i] = make_frag(ap[hl], ap[2 + hl]);
            // B fragment: lane<16 holds K{0..7,16..23} of column l16; lane>=16 K{8..15,24..31}
            const uint4* bp = (const uint4*)&lw[((nj0 + i) * 16 + l16) * 32];
            bfr[i] = make_frag(bp[hl], bp[2 + hl]);
        }
        #pragma unroll
        for (int i = 0; i < 2; ++i)
            #pragma unroll
            for (int j = 0; j < 2; ++j)
                acc[i][j] = __builtin_amdgcn_wmma_f32_16x16x32_bf16(
                    false, af[i], false, bfr[j], (short)0, acc[i][j], false, false);
        __syncthreads();
    }

    // epilogue: bias + store.  D-layout: VGPR r -> row +r (+8 for hi lanes), lane -> col
    #pragma unroll
    for (int i = 0; i < 2; ++i) {
        #pragma unroll
        for (int j = 0; j < 2; ++j) {
            const int ng = tileN + (nj0 + j) * 16 + l16;
            const float bv = bias[ng];
            #pragma unroll
            for (int r = 0; r < 8; ++r) {
                const int mg = tileM + (mi0 + i) * 16 + r + 8 * hl;
                const float val = acc[i][j][r] + bv;
                if (MODE == 0) {
                    ((float*)out)[(size_t)mg * DIM_D + ng] = val;
                } else {
                    const int bb = mg >> 11, s = mg & (DIM_S - 1);
                    const int h  = ng >> 7,  hd = ng & (DIM_HD - 1);
                    if (MODE == 1)
                        ((unsigned short*)out)[(((bb * DIM_H + h) * DIM_S) + s) * DIM_HD + hd] = f32_bf16(val);
                    else
                        ((unsigned short*)out)[(((bb * DIM_H + h) * DIM_HD) + hd) * DIM_S + s] = f32_bf16(val);
                }
            }
        }
    }
}

// ---------------------------------------------------------------------------
// Flash attention: 1 wave per (b,h,qtile16).  Q/K bf16 [bh, s, hd],
// V bf16 transposed [bh, hd, s].  Online softmax in f32, bf16 out [b,s,D].
// ---------------------------------------------------------------------------
__global__ __launch_bounds__(32)
void flash_attn_kernel(const unsigned short* __restrict__ qm,
                       const unsigned short* __restrict__ km,
                       const unsigned short* __restrict__ vtm,
                       unsigned short* __restrict__ om)
{
    __shared__ unsigned short lp[16 * 32];   // P tile (16 q x 32 keys) bf16

    const int lane  = threadIdx.x & 31;
    const int hl    = lane >> 4;
    const int l16   = lane & 15;
    const int bh    = blockIdx.x >> 7;       // 0..31  (b*H + h)
    const int qt    = blockIdx.x & 127;
    const int qbase = qt * 16;

    // Q A-fragments for all 4 head-dim chunks, kept in registers
    const uint4* qrow = (const uint4*)(qm + (size_t)(bh * DIM_S + qbase + l16) * DIM_HD);
    v16bf qfrag[4];
    #pragma unroll
    for (int c = 0; c < 4; ++c)
        qfrag[c] = make_frag(qrow[4 * c + hl], qrow[4 * c + 2 + hl]);

    float mrun[8], lrun[8];
    v8f oacc[8];
    #pragma unroll
    for (int r = 0; r < 8; ++r) { mrun[r] = NEG_BIG; lrun[r] = 0.0f; }
    #pragma unroll
    for (int dc = 0; dc < 8; ++dc) oacc[dc] = v8f{};

    const int ntile = ((qbase + 15) >> 5) + 1;   // causal: 32-key tiles needed

    for (int t32 = 0; t32 < ntile; ++t32) {
        const int kbase = t32 * 32;

        // ---- scores: two 16q x 16key tiles, K-loop over head dim ----
        v8f sc[2];
        sc[0] = v8f{}; sc[1] = v8f{};
        #pragma unroll
        for (int tt = 0; tt < 2; ++tt) {
            const int key = kbase + tt * 16 + l16;
            const uint4* krow = (const uint4*)(km + (size_t)(bh * DIM_S + key) * DIM_HD);
            #pragma unroll
            for (int c = 0; c < 4; ++c) {
                v16bf kf = make_frag(krow[4 * c + hl], krow[4 * c + 2 + hl]);
                sc[tt] = __builtin_amdgcn_wmma_f32_16x16x32_bf16(
                    false, qfrag[c], false, kf, (short)0, sc[tt], false, false);
            }
        }

        // ---- scale + causal mask + online softmax ----
        float alpha[8];
        #pragma unroll
        for (int r = 0; r < 8; ++r) {
            const int qg = qbase + r + 8 * hl;
            #pragma unroll
            for (int tt = 0; tt < 2; ++tt) {
                const int kg = kbase + tt * 16 + l16;
                float sv = sc[tt][r] * ATT_SCALE;
                sc[tt][r] = (kg <= qg) ? sv : NEG_BIG;
            }
            float mr = fmaxf(sc[0][r], sc[1][r]);
            #pragma unroll
            for (int msk = 1; msk < 16; msk <<= 1)
                mr = fmaxf(mr, __shfl_xor(mr, msk, 32));
            const float newm = fmaxf(mrun[r], mr);

            const float p0 = __expf(sc[0][r] - newm);
            const float p1 = __expf(sc[1][r] - newm);
            const int m = r + 8 * hl;
            lp[m * 32 + l16]      = f32_bf16(p0);
            lp[m * 32 + 16 + l16] = f32_bf16(p1);

            float ps = p0 + p1;
            #pragma unroll
            for (int msk = 1; msk < 16; msk <<= 1)
                ps += __shfl_xor(ps, msk, 32);

            alpha[r] = __expf(mrun[r] - newm);
            lrun[r]  = lrun[r] * alpha[r] + ps;
            mrun[r]  = newm;
        }
        #pragma unroll
        for (int dc = 0; dc < 8; ++dc)
            #pragma unroll
            for (int r = 0; r < 8; ++r)
                oacc[dc][r] *= alpha[r];

        __syncthreads();   // P visible (D-layout -> A-layout via LDS)

        // P A-fragment (16 x 32 keys)
        const uint4* pr = (const uint4*)&lp[l16 * 32];
        v16bf pf = make_frag(pr[hl], pr[2 + hl]);

        // ---- O += P x V : V^T layout makes B-fragments contiguous ----
        #pragma unroll
        for (int dc = 0; dc < 8; ++dc) {
            const int d = dc * 16 + l16;
            const uint4* vp = (const uint4*)(vtm + (size_t)(bh * DIM_HD + d) * DIM_S + kbase);
            v16bf vf = make_frag(vp[hl], vp[2 + hl]);
            oacc[dc] = __builtin_amdgcn_wmma_f32_16x16x32_bf16(
                false, pf, false, vf, (short)0, oacc[dc], false, false);
        }
        __syncthreads();   // protect lp before next tile's stores
    }

    // ---- finalize: divide by row sums, bf16 store at [b, s, h*HD + d] ----
    const int b = bh >> 4, h = bh & 15;
    #pragma unroll
    for (int r = 0; r < 8; ++r) {
        const int sidx = qbase + r + 8 * hl;
        const float inv = 1.0f / lrun[r];
        unsigned short* orow = om + (size_t)(b * DIM_S + sidx) * DIM_D + h * DIM_HD;
        #pragma unroll
        for (int dc = 0; dc < 8; ++dc)
            orow[dc * 16 + l16] = f32_bf16(oacc[dc][r] * inv);
    }
}

// ---------------------------------------------------------------------------
extern "C" void kernel_launch(void* const* d_in, const int* in_sizes, int n_in,
                              void* d_out, int out_size, void* d_ws, size_t ws_size,
                              hipStream_t stream)
{
    const float* x   = (const float*)d_in[0];
    const float* q_w = (const float*)d_in[1];
    const float* q_b = (const float*)d_in[2];
    const float* k_w = (const float*)d_in[3];
    const float* k_b = (const float*)d_in[4];
    const float* v_w = (const float*)d_in[5];
    const float* v_b = (const float*)d_in[6];
    const float* o_w = (const float*)d_in[7];
    const float* o_b = (const float*)d_in[8];

    const size_t NM = (size_t)DIM_M * DIM_D;   // 8.39M
    const size_t NW = (size_t)DIM_D * DIM_D;   // 4.19M
    unsigned short* xb  = (unsigned short*)d_ws;
    unsigned short* qwb = xb  + NM;
    unsigned short* kwb = qwb + NW;
    unsigned short* vwb = kwb + NW;
    unsigned short* owb = vwb + NW;
    unsigned short* qws = owb + NW;
    unsigned short* kws = qws + NM;
    unsigned short* vws = kws + NM;
    unsigned short* aws = vws + NM;

    // pass 0: one-shot bf16 conversion (removes all per-tile convert VALU)
    cvt_bf16_kernel<<<dim3((int)(NM / 8 / 256)), 256, 0, stream>>>(x,   xb,  (int)(NM / 8));
    cvt_bf16_kernel<<<dim3((int)(NW / 8 / 256)), 256, 0, stream>>>(q_w, qwb, (int)(NW / 8));
    cvt_bf16_kernel<<<dim3((int)(NW / 8 / 256)), 256, 0, stream>>>(k_w, kwb, (int)(NW / 8));
    cvt_bf16_kernel<<<dim3((int)(NW / 8 / 256)), 256, 0, stream>>>(v_w, vwb, (int)(NW / 8));
    cvt_bf16_kernel<<<dim3((int)(NW / 8 / 256)), 256, 0, stream>>>(o_w, owb, (int)(NW / 8));

    dim3 grid(DIM_D / 64, DIM_M / 128);   // (32, 32)
    gemm_bias_kernel<1><<<grid, 256, 0, stream>>>(xb, qwb, q_b, qws);
    gemm_bias_kernel<1><<<grid, 256, 0, stream>>>(xb, kwb, k_b, kws);
    gemm_bias_kernel<2><<<grid, 256, 0, stream>>>(xb, vwb, v_b, vws);

    flash_attn_kernel<<<dim3(32 * 128), 32, 0, stream>>>(qws, kws, vws, aws);

    gemm_bias_kernel<0><<<grid, 256, 0, stream>>>(aws, owb, o_b, d_out);
}
